// moments_layer_59811714564353
// MI455X (gfx1250) — matrix operand order repaired and checked by reading, not compiled, not run on hardware
//
#include <hip/hip_runtime.h>
#include <math.h>

// ---------------------------------------------------------------- constants
#define B_N   65536
#define K_N   16
#define D_N   8
#define NF    80            // padded feature count (73 used: 8 dsel + 64 p2f + 1)
#define GSTRIDE (NF * NF)   // 6400 floats per cluster Gram
#define CHUNKS 32           // sample-chunks per cluster in Gram phase
#define SPF   128           // samples per LDS fill (4 WMMA K-steps of 32)
#define LDSP  136           // uT row stride in halves (272B: 16B-aligned, odd*16 banks)

typedef __attribute__((ext_vector_type(16))) _Float16 v16h;
typedef __attribute__((ext_vector_type(8)))  _Float16 v8h;
typedef __attribute__((ext_vector_type(8)))  float    v8f;

// ---------------------------------------------------------------- phase 1
// resp (softmax over -0.5*dist^2), hard assignment, cluster histogram
__global__ void __launch_bounds__(256)
k_resp_assign(const float* __restrict__ x, const float* __restrict__ centers,
              float* __restrict__ resp, int* __restrict__ assign,
              int* __restrict__ counts) {
  __shared__ float csh[K_N * D_N];
  __shared__ int   hist[K_N];
  const int tid = threadIdx.x;
  if (tid < K_N * D_N) csh[tid] = centers[tid];
  if (tid < K_N) hist[tid] = 0;
  __syncthreads();

  const int b = blockIdx.x * 256 + tid;   // grid exactly covers B_N
  const float4* xp = (const float4*)(x + (size_t)b * D_N);
  float4 x0 = xp[0], x1 = xp[1];
  float xv[D_N] = {x0.x, x0.y, x0.z, x0.w, x1.x, x1.y, x1.z, x1.w};

  float logit[K_N];
  float bestd = 3.4e38f;
  int   bi = 0;
  #pragma unroll
  for (int k = 0; k < K_N; ++k) {
    float dist = 0.f;
    #pragma unroll
    for (int d = 0; d < D_N; ++d) {
      float df = xv[d] - csh[k * D_N + d];
      dist += df * df;
    }
    logit[k] = -0.5f * dist;
    if (dist < bestd) { bestd = dist; bi = k; }   // first-min tie break
  }
  float mx = logit[0];
  #pragma unroll
  for (int k = 1; k < K_N; ++k) mx = fmaxf(mx, logit[k]);
  float s = 0.f;
  #pragma unroll
  for (int k = 0; k < K_N; ++k) { float e = expf(logit[k] - mx); logit[k] = e; s += e; }
  const float invs = 1.f / s;

  float4* op = (float4*)(resp + (size_t)b * K_N);
  #pragma unroll
  for (int q = 0; q < 4; ++q) {
    float4 o;
    o.x = logit[q * 4 + 0] * invs; o.y = logit[q * 4 + 1] * invs;
    o.z = logit[q * 4 + 2] * invs; o.w = logit[q * 4 + 3] * invs;
    op[q] = o;
  }
  assign[b] = bi;
  atomicAdd(&hist[bi], 1);
  __syncthreads();
  if (tid < K_N) atomicAdd(&counts[tid], hist[tid]);
}

// ---------------------------------------------------------------- phase 2
__global__ void k_scan(const int* __restrict__ counts,
                       int* __restrict__ base, int* __restrict__ cursor) {
  if (threadIdx.x == 0) {
    int run = 0;
    for (int k = 0; k < K_N; ++k) { base[k] = run; cursor[k] = run; run += counts[k]; }
  }
}

// ---------------------------------------------------------------- phase 3
__global__ void __launch_bounds__(256)
k_scatter(const int* __restrict__ assign, int* __restrict__ cursor,
          int* __restrict__ perm) {
  const int b = blockIdx.x * 256 + threadIdx.x;
  const int k = assign[b];
  const int pos = atomicAdd(&cursor[k], 1);
  perm[pos] = b;
}

// ---------------------------------------------------------------- phase 4
// Upper-triangle tile index (0..14) -> (ti, tj) of the 5x5 grid, ti <= tj.
__device__ __forceinline__ void tri_map(int t, int& ti, int& tj) {
  int bb = 0, i = 0;
  #pragma unroll
  for (int r = 0; r < 5; ++r) {
    const int w = 5 - r;
    if (t >= bb + w) { bb += w; i = r + 1; }
  }
  ti = i; tj = i + (t - bb);
}

__device__ __forceinline__ v16h frag_a(const _Float16* row, int sb, int kbA) {
  // A 16x32 f16 layout: lane<16 holds K in {kbA..kbA+7} U {kbA+16..kbA+23}
  v8h lo = *(const v8h*)(row + sb + kbA);
  v8h hi = *(const v8h*)(row + sb + kbA + 16);
  v16h r;
  #pragma unroll
  for (int i = 0; i < 8; ++i) { r[i] = lo[i]; r[i + 8] = hi[i]; }
  return r;
}

__device__ __forceinline__ v16h frag_b(const _Float16* row, int sb, int kbB) {
  // B 32x16 f16 layout: lane<16 holds K in {kbB..kbB+7} U {kbB+16..kbB+23}
  v8h lo = *(const v8h*)(row + sb + kbB);
  v8h hi = *(const v8h*)(row + sb + kbB + 16);
  v16h r;
  #pragma unroll
  for (int i = 0; i < 8; ++i) { r[i] = lo[i]; r[i + 8] = hi[i]; }
  return r;
}

// Per-cluster Gram of augmented feature rows u = [dsel(8), dsel⊗dsel(64), 1, 0..]
// via v_wmma_f32_16x16x32_f16.  uT is feature-major [NF][SPF] f16 in LDS so A and
// B fragments are contiguous ds_load_b128 pairs.  8 waves x 2 fixed upper-triangle
// tiles -> branchless WMMA with full EXEC; symmetric half written at flush time.
__global__ void __launch_bounds__(256)
k_gram(const float* __restrict__ x, const float* __restrict__ centers,
       const int* __restrict__ base, const int* __restrict__ counts,
       const int* __restrict__ perm, float* __restrict__ G) {
  __shared__ _Float16 uT[NF][LDSP];
  __shared__ float dsh[SPF][D_N];
  __shared__ float wsh[SPF];

  const int k     = blockIdx.x / CHUNKS;
  const int chunk = blockIdx.x % CHUNKS;
  const int tid   = threadIdx.x;
  const int lane  = tid & 31;
  const int wave  = tid >> 5;
  const int half  = lane >> 4;
  const int m     = lane & 15;

  const int cbase = base[k];
  const int cnt   = counts[k];
  float ck[D_N];
  #pragma unroll
  for (int d = 0; d < D_N; ++d) ck[d] = centers[k * D_N + d];

  // Fixed tile assignment: wave w owns triangle tiles w and 8+w (wave7's second
  // slot duplicates tile 14 and is never flushed).
  const int  t0     = wave;
  const bool flush1 = (8 + wave) < 15;
  const int  t1     = flush1 ? (8 + wave) : 14;
  int ti0, tj0, ti1, tj1;
  tri_map(t0, ti0, tj0);
  tri_map(t1, ti1, tj1);

  const _Float16* a0row = &uT[ti0 * 16 + m][0];
  const _Float16* b0row = &uT[tj0 * 16 + m][0];
  const _Float16* a1row = &uT[ti1 * 16 + m][0];
  const _Float16* b1row = &uT[tj1 * 16 + m][0];
  const int kbA = half * 8;
  const int kbB = half * 8;

  v8f acc0, acc1;
  #pragma unroll
  for (int i = 0; i < 8; ++i) { acc0[i] = 0.f; acc1[i] = 0.f; }

  for (int off = chunk * SPF; off < cnt; off += CHUNKS * SPF) {
    // ---- build dsel rows for SPF samples
    if (tid < SPF) {
      const int i = off + tid;
      float dv[D_N];
      float w = 0.f;
      if (i < cnt) {
        const int b = perm[cbase + i];
        const float4* xp = (const float4*)(x + (size_t)b * D_N);
        float4 x0 = xp[0], x1 = xp[1];
        dv[0] = x0.x - ck[0]; dv[1] = x0.y - ck[1];
        dv[2] = x0.z - ck[2]; dv[3] = x0.w - ck[3];
        dv[4] = x1.x - ck[4]; dv[5] = x1.y - ck[5];
        dv[6] = x1.z - ck[6]; dv[7] = x1.w - ck[7];
        w = 1.f;
      } else {
        #pragma unroll
        for (int d = 0; d < D_N; ++d) dv[d] = 0.f;
      }
      #pragma unroll
      for (int d = 0; d < D_N; ++d) dsh[tid][d] = dv[d];
      wsh[tid] = w;
    }
    __syncthreads();

    // ---- expand to NF f16 features, feature-major (2 threads per sample)
    {
      const int g = tid >> 7;          // 0..1
      const int s = tid & (SPF - 1);   // 0..127
      float dv[D_N];
      #pragma unroll
      for (int d = 0; d < D_N; ++d) dv[d] = dsh[s][d];
      const float w = wsh[s];
      #pragma unroll
      for (int ff = 0; ff < NF / 2; ++ff) {
        const int f = g * (NF / 2) + ff;
        float val;
        if (f < 8)        val = dv[f];
        else if (f < 72)  { int i = (f - 8) >> 3, j = (f - 8) & 7; val = dv[i] * dv[j]; }
        else if (f == 72) val = w;          // count feature (0 for padding rows)
        else              val = 0.f;        // zero padding features 73..79
        uT[f][s] = (_Float16)val;
      }
    }
    __syncthreads();

    // ---- 4 unrolled WMMA K-steps over the 128-sample fill, 2 tiles per wave
    #pragma unroll
    for (int kc = 0; kc < SPF / 32; ++kc) {
      const int sb = kc * 32;
      v16h a0 = frag_a(a0row, sb, kbA);
      v16h c0 = frag_b(b0row, sb, kbB);
      acc0 = __builtin_amdgcn_wmma_f32_16x16x32_f16(
          false, a0, false, c0, (short)0, acc0, false, false);
      v16h a1 = frag_a(a1row, sb, kbA);
      v16h c1 = frag_b(b1row, sb, kbB);
      acc1 = __builtin_amdgcn_wmma_f32_16x16x32_f16(
          false, a1, false, c1, (short)0, acc1, false, false);
    }
    __syncthreads();
  }

  // ---- flush (C/D layout: VGPR r, lanes0-15 M=r, lanes16-31 M=8+r), mirror
  float* Gk = G + (size_t)k * GSTRIDE;
  #pragma unroll
  for (int r = 0; r < 8; ++r) {
    const int row = ti0 * 16 + half * 8 + r;
    const int col = tj0 * 16 + m;
    atomicAdd(&Gk[row * NF + col], acc0[r]);
    if (ti0 != tj0) atomicAdd(&Gk[col * NF + row], acc0[r]);
  }
  if (flush1) {
    #pragma unroll
    for (int r = 0; r < 8; ++r) {
      const int row = ti1 * 16 + half * 8 + r;
      const int col = tj1 * 16 + m;
      atomicAdd(&Gk[row * NF + col], acc1[r]);
      if (ti1 != tj1) atomicAdd(&Gk[col * NF + row], acc1[r]);
    }
  }
}

// ---------------------------------------------------------------- phase 5
__device__ __forceinline__ float sgnp(float m) { return m < 0.f ? -1.f : 1.f; }
__device__ __forceinline__ float shr2(float m) {
  return sgnp(m) * (sqrtf(fabsf(m) + 0.25f) - 0.5f);
}
__device__ __forceinline__ float shr3(float m) {
  return sgnp(m) * (cbrtf(fabsf(m) + 0.19245008973f) - 0.57735026919f);
}
__device__ __forceinline__ float shr4(float m) {
  return sgnp(m) * (sqrtf(sqrtf(fabsf(m) + 0.15749013123f)) - 0.62996052494f);
}

__global__ void __launch_bounds__(256)
k_penalty(const float* __restrict__ G, const float* __restrict__ mw1,
          const float* __restrict__ t1, const float* __restrict__ t2,
          const float* __restrict__ t3, const float* __restrict__ t4,
          float* __restrict__ out) {
  const int k = blockIdx.x;
  const float* Gk = G + (size_t)k * GSTRIDE;
  const float cw  = Gk[72 * NF + 72];          // Σ 1-feature == cluster count
  const float cwn = cw / (float)B_N;           // Σcw == B (every sample assigned)
  const float inv = 1.f / (cw + 1e-7f);

  float local = 0.f;
  for (int idx = threadIdx.x; idx < NF * NF; idx += 256) {
    const int r = idx / NF, c = idx % NF;
    const float gv = Gk[idx];
    if (r == 72 && c < 8) {                         // m1 (count-normalized)
      const float d = gv * inv - t1[c];
      local += 1.0f * cwn * mw1[c] * d * d;
    } else if (r < 8 && c < 8) {                    // m2 (count-normalized)
      const float d = shr2(gv * inv) - shr2(t2[r * 8 + c]);
      local += 0.5f * cwn * mw1[c] * d * d;
    } else if (r >= 8 && r < 72 && c < 8) {         // m3 raw sum
      const float d = shr3(gv) - shr3(t3[(r - 8) * 8 + c]);
      local += 0.25f * cwn * mw1[c] * d * d;
    } else if (r >= 8 && r < 72 && c >= 8 && c < 72) {  // m4 raw sum
      const float d = shr4(gv) - shr4(t4[(r - 8) * 64 + (c - 8)]);
      local += 0.125f * cwn * mw1[(c - 8) & 7] * d * d;
    }
  }
  __shared__ float red[256];
  red[threadIdx.x] = local;
  __syncthreads();
  for (int st = 128; st > 0; st >>= 1) {
    if (threadIdx.x < st) red[threadIdx.x] += red[threadIdx.x + st];
    __syncthreads();
  }
  if (threadIdx.x == 0) atomicAdd(out, red[0]);
}

// ---------------------------------------------------------------- launcher
extern "C" void kernel_launch(void* const* d_in, const int* in_sizes, int n_in,
                              void* d_out, int out_size, void* d_ws, size_t ws_size,
                              hipStream_t stream) {
  (void)in_sizes; (void)n_in; (void)out_size; (void)ws_size;
  const float* x       = (const float*)d_in[0];
  const float* centers = (const float*)d_in[1];
  const float* mw1     = (const float*)d_in[2];
  const float* t1      = (const float*)d_in[3];
  const float* t2      = (const float*)d_in[4];
  const float* t3      = (const float*)d_in[5];
  const float* t4      = (const float*)d_in[6];

  float* resp    = (float*)d_out;
  float* penalty = resp + (size_t)B_N * K_N;

  char* ws = (char*)d_ws;
  int* counts = (int*)ws;  ws += 16 * sizeof(int);
  int* base   = (int*)ws;  ws += 16 * sizeof(int);
  int* cursor = (int*)ws;  ws += 16 * sizeof(int);
  int* assign = (int*)ws;  ws += (size_t)B_N * sizeof(int);
  int* perm   = (int*)ws;  ws += (size_t)B_N * sizeof(int);
  float* G    = (float*)ws;   // 16 * 6400 floats (offset is 16B aligned)

  hipMemsetAsync(counts, 0, 16 * sizeof(int), stream);
  hipMemsetAsync(G, 0, (size_t)K_N * GSTRIDE * sizeof(float), stream);
  hipMemsetAsync(penalty, 0, sizeof(float), stream);

  k_resp_assign<<<B_N / 256, 256, 0, stream>>>(x, centers, resp, assign, counts);
  k_scan<<<1, 32, 0, stream>>>(counts, base, cursor);
  k_scatter<<<B_N / 256, 256, 0, stream>>>(assign, cursor, perm);
  k_gram<<<K_N * CHUNKS, 256, 0, stream>>>(x, centers, base, counts, perm, G);
  k_penalty<<<K_N, 256, 0, stream>>>(G, mw1, t1, t2, t3, t4, penalty);
}
